// SelfAttention_75204877353514
// MI455X (gfx1250) — matrix operand order repaired and checked by reading, not compiled, not run on hardware
//
#include <hip/hip_runtime.h>

// ---------------------------------------------------------------------------
// Self-attention forward for gfx1250 (CDNA5), bf16 WMMA pipeline.
//   bs=16, n=1024, d=256, h=8.  All matrix math via v_wmma_f32_16x16x32_bf16.
// Outputs (f32, concatenated): out (16,1024,256) then probs (16,8,1024,1024).
// Workspace layout (227 MB):
//   xn_f32 (16MB) | xn_bf16 (8MB) | xnT_bf16 (8MB) |
//   Wq_bf16 (1MB) | Wk_bf16 (1MB) | Wv_bf16 (1MB)  |
//   Q_bf16 (64MB) | K_bf16 (64MB) | ctx_bf16 (64MB)
// ---------------------------------------------------------------------------

#define BS    16
#define SEQ   1024
#define DIM   256
#define HEADS 8
#define HD    2048          // HEADS*DIM

typedef __attribute__((ext_vector_type(16))) __bf16 v16bf;
typedef __attribute__((ext_vector_type(8)))  __bf16 v8bf;
typedef __attribute__((ext_vector_type(8)))  float  v8f;

// float -> bf16, round-to-nearest-even (bit manipulation: host+device safe)
__device__ __forceinline__ __bf16 f2bf(float f) {
  union { float f; unsigned u; } v; v.f = f;
  unsigned r = (v.u + 0x7FFFu + ((v.u >> 16) & 1u)) >> 16;
  unsigned short s = (unsigned short)r;
  __bf16 b;
  __builtin_memcpy(&b, &s, 2);
  return b;
}

// A-matrix fragment, 16x32 bf16 (M x K).  K order used here (not the ISA table order):
//   lane L: row m = L&15, K-half = L>>4; holds K = kb + half*16 + [0..15]
//   contiguous (same K order as load_b32x16).
__device__ __forceinline__ v16bf load_a16x32(const __bf16* base, int ld,
                                             int kb, int lane) {
  const __bf16* p = base + (size_t)(lane & 15) * ld + kb + ((lane >> 4) << 4);
  return *(const v16bf*)p;            // 32B load
}

// B-matrix fragment, 32x16 bf16 (K x N).  K order used here (same as
// load_a16x32): lane L: col n = L&15, K-half = L>>4; holds
// K = kb + half*16 + [0..15] contiguous.  Column n of B is a contiguous
// K-run of source row n, i.e. this computes X * W^T naturally.
__device__ __forceinline__ v16bf load_b32x16(const __bf16* base, int ld,
                                             int kb, int lane) {
  const __bf16* p = base + (size_t)(lane & 15) * ld + kb + ((lane >> 4) << 4);
  return *(const v16bf*)p;            // 32B (2 x b128) load
}

__device__ __forceinline__ v8f wmma_bf16(v16bf a, v16bf b, v8f c) {
  // (neg_a, A, neg_b, B, c_mod, C, reuse_a, reuse_b)
  return __builtin_amdgcn_wmma_f32_16x16x32_bf16(false, a, false, b,
                                                 (short)0, c, false, false);
}

// Store a 16x16 f32 D-tile as bf16.  D layout: row = r + 8*(lane>=16), col = lane&15.
__device__ __forceinline__ void store_tile_bf16(__bf16* Y, size_t ld, int m0,
                                                int o0, v8f acc, int lane) {
  const int n = lane & 15, hh = lane >> 4;
#pragma unroll
  for (int r = 0; r < 8; ++r)
    Y[(size_t)(m0 + r + hh * 8) * ld + o0 + n] = f2bf(acc[r]);
}

// ---------------------------------------------------------------------------
// K1: pre-norm LayerNorm.  One 256-thread block per token row.
//     Emits xn (f32), xn (bf16), and per-batch transposed xnT (bf16, D x N).
// ---------------------------------------------------------------------------
__global__ __launch_bounds__(256) void ln_kernel(
    const float* __restrict__ x, const float* __restrict__ gamma,
    const float* __restrict__ beta, float* __restrict__ xn,
    __bf16* __restrict__ xnb, __bf16* __restrict__ xnT) {
  __shared__ float red[256];
  const int row = blockIdx.x;           // 0..16383 = b*1024 + n
  const int t   = threadIdx.x;          // 0..255
  const float v = x[(size_t)row * DIM + t];

  red[t] = v; __syncthreads();
  for (int s = 128; s > 0; s >>= 1) {
    if (t < s) red[t] += red[t + s];
    __syncthreads();
  }
  const float mu = red[0] * (1.0f / DIM);
  __syncthreads();

  const float dv = v - mu;
  red[t] = dv * dv; __syncthreads();
  for (int s = 128; s > 0; s >>= 1) {
    if (t < s) red[t] += red[t + s];
    __syncthreads();
  }
  const float var = red[0] * (1.0f / DIM);
  const float y = dv * rsqrtf(var + 1e-5f) * gamma[t] + beta[t];

  xn[(size_t)row * DIM + t] = y;
  const __bf16 yb = f2bf(y);
  xnb[(size_t)row * DIM + t] = yb;
  const int b = row >> 10, n = row & 1023;
  xnT[((size_t)b * DIM + t) * SEQ + n] = yb;   // V^T for the P*V GEMM
}

// ---------------------------------------------------------------------------
// K2: convert Wq/Wk/Wv to bf16 (all three are 524288 elements).
// ---------------------------------------------------------------------------
__global__ __launch_bounds__(256) void wcvt_kernel(
    const float* __restrict__ wq, const float* __restrict__ wk,
    const float* __restrict__ wv, __bf16* __restrict__ wqb,
    __bf16* __restrict__ wkb, __bf16* __restrict__ wvb) {
  const size_t i = (size_t)blockIdx.x * 256 + threadIdx.x;
  wqb[i] = f2bf(wq[i]);
  wkb[i] = f2bf(wk[i]);
  wvb[i] = f2bf(wv[i]);
}

// ---------------------------------------------------------------------------
// K3: Q/K projection.  Y = xn (16384 x 256) @ W^T (256 x 2048), bf16 out.
//     2x2 register blocking: each wave owns a 32x32 output tile (4 accs);
//     per k-step: 2 A-frags + 2 B-frags -> 4 WMMAs (halves L0 traffic/wmma).
//     grid (512 row tiles of 32, 16 col groups of 128, 2 {Q,K}); 4 waves.
// ---------------------------------------------------------------------------
__global__ __launch_bounds__(128) void proj_qk_kernel(
    const __bf16* __restrict__ xnb, const __bf16* __restrict__ wqb,
    const __bf16* __restrict__ wkb, __bf16* __restrict__ qb,
    __bf16* __restrict__ kb) {
  const int lane = threadIdx.x & 31;
  const int w    = threadIdx.x >> 5;
  const int m0   = blockIdx.x * 32;
  const int o0   = blockIdx.y * 128 + w * 32;
  const __bf16* W = blockIdx.z ? wkb : wqb;
  __bf16*       Y = blockIdx.z ? kb  : qb;

  const __bf16* a0base = xnb + (size_t)m0 * DIM;
  const __bf16* a1base = a0base + (size_t)16 * DIM;
  const __bf16* b0base = W + (size_t)o0 * DIM;
  const __bf16* b1base = b0base + (size_t)16 * DIM;

  v8f acc00 = {}, acc01 = {}, acc10 = {}, acc11 = {};
#pragma unroll
  for (int k = 0; k < DIM / 32; ++k) {
    const int kb_ = k * 32;
    v16bf a0 = load_a16x32(a0base, DIM, kb_, lane);
    v16bf a1 = load_a16x32(a1base, DIM, kb_, lane);
    v16bf b0 = load_b32x16(b0base, DIM, kb_, lane);
    v16bf b1 = load_b32x16(b1base, DIM, kb_, lane);
    acc00 = wmma_bf16(a0, b0, acc00);
    acc01 = wmma_bf16(a0, b1, acc01);
    acc10 = wmma_bf16(a1, b0, acc10);
    acc11 = wmma_bf16(a1, b1, acc11);
  }

  store_tile_bf16(Y, HD, m0,      o0,      acc00, lane);
  store_tile_bf16(Y, HD, m0,      o0 + 16, acc01, lane);
  store_tile_bf16(Y, HD, m0 + 16, o0,      acc10, lane);
  store_tile_bf16(Y, HD, m0 + 16, o0 + 16, acc11, lane);
}

// ---------------------------------------------------------------------------
// K4: attention core.  One block (4 waves) per (b, h, 16-row query tile).
//     Scores 16x1024 held in LDS (f32), full softmax, probs written f32 to
//     d_out (coalesced) + bf16 to LDS, then context = P @ V via WMMA where
//     V = xn (shared across heads), read from the transposed xnT copy.
// ---------------------------------------------------------------------------
__global__ __launch_bounds__(128) void attn_kernel(
    const __bf16* __restrict__ qb, const __bf16* __restrict__ kb,
    const __bf16* __restrict__ xnT, __bf16* __restrict__ ctx,
    float* __restrict__ probs) {
  extern __shared__ char smem[];
  float*  s_sc   = (float*)smem;                                // 16*1024 f32
  __bf16* s_p    = (__bf16*)(smem + 16 * SEQ * 4);              // 16*1024 bf16
  float*  s_part = (float*)(smem + 16 * SEQ * 4 + 16 * SEQ * 2);// 128 f32

  const int tid  = threadIdx.x;
  const int lane = tid & 31;
  const int w    = tid >> 5;
  const int i0   = blockIdx.x * 16;   // query tile
  const int h    = blockIdx.y;
  const int b    = blockIdx.z;

  // ---- Phase A: scores S = (Q K^T) * 1/sqrt(d).  Wave w owns cols [w*256,+256)
  const __bf16* qbase = qb + ((size_t)b * SEQ + i0) * HD + h * DIM;
  v16bf afr[8];
#pragma unroll
  for (int k = 0; k < 8; ++k) afr[k] = load_a16x32(qbase, HD, k * 32, lane);

  const float scale = 0.0625f;  // 1/sqrt(256)
  for (int jt = 0; jt < 16; ++jt) {
    const int j0 = w * 256 + jt * 16;
    const __bf16* kbase = kb + ((size_t)b * SEQ + j0) * HD + h * DIM;
    __builtin_prefetch(kbase + 16 * HD, 0, 0);    // global_prefetch next tile
    v8f acc = {};
#pragma unroll
    for (int k = 0; k < 8; ++k) {
      v16bf bf = load_b32x16(kbase, HD, k * 32, lane);
      acc = wmma_bf16(afr[k], bf, acc);
    }
    const int n = lane & 15, hh = lane >> 4;
#pragma unroll
    for (int r = 0; r < 8; ++r)
      s_sc[(r + hh * 8) * SEQ + j0 + n] = acc[r] * scale;
  }
  __syncthreads();

  // ---- Phase B: row softmax; write probs (f32, coalesced) + bf16 to LDS.
  float* prow = probs + (((size_t)b * HEADS + h) * SEQ + i0) * SEQ;
  for (int r = 0; r < 16; ++r) {
    float mx = -3.402823466e38f;
    for (int j = tid; j < SEQ; j += 128) mx = fmaxf(mx, s_sc[r * SEQ + j]);
    s_part[tid] = mx; __syncthreads();
    for (int s = 64; s > 0; s >>= 1) {
      if (tid < s) s_part[tid] = fmaxf(s_part[tid], s_part[tid + s]);
      __syncthreads();
    }
    const float rmax = s_part[0];
    __syncthreads();

    float sum = 0.f;
    for (int j = tid; j < SEQ; j += 128) {
      const float e = __expf(s_sc[r * SEQ + j] - rmax);
      s_sc[r * SEQ + j] = e;
      sum += e;
    }
    s_part[tid] = sum; __syncthreads();
    for (int s = 64; s > 0; s >>= 1) {
      if (tid < s) s_part[tid] += s_part[tid + s];
      __syncthreads();
    }
    const float inv = 1.f / s_part[0];
    __syncthreads();

    for (int j = tid; j < SEQ; j += 128) {
      const float p = s_sc[r * SEQ + j] * inv;
      prow[(size_t)r * SEQ + j] = p;
      s_p[r * SEQ + j] = f2bf(p);
    }
  }
  __syncthreads();

  // ---- Phase C: context = P (16x1024) @ V (1024x256).  Wave w owns d-cols
  //      [w*64, +64) = 4 WMMA tiles; K loop 1024/32 = 32, A fragment reused.
  const __bf16* vbase = xnT + (size_t)b * DIM * SEQ;   // [d][n], contiguous K
  v8f cacc[4];
#pragma unroll
  for (int jt = 0; jt < 4; ++jt) cacc[jt] = {};

  for (int kk = 0; kk < SEQ / 32; ++kk) {
    v16bf a = load_a16x32(s_p, SEQ, kk * 32, lane);    // from LDS
#pragma unroll
    for (int jt = 0; jt < 4; ++jt) {
      const int c0 = w * 64 + jt * 16;
      v16bf bf = load_b32x16(vbase + (size_t)c0 * SEQ, SEQ, kk * 32, lane);
      cacc[jt] = wmma_bf16(a, bf, cacc[jt]);
    }
  }

#pragma unroll
  for (int jt = 0; jt < 4; ++jt) {
    const int c0 = w * 64 + jt * 16;
    store_tile_bf16(ctx + (size_t)b * SEQ * HD + h * DIM, HD, i0, c0,
                    cacc[jt], lane);
  }
}

// ---------------------------------------------------------------------------
// K5: out = ctx (16384 x 2048) @ Wv^T (2048 x 256) + xn (residual), f32 out.
//     2x2 register blocking (32x32 per wave), K = 64 steps.
//     grid (512 row tiles of 32, 2 col groups of 128); 4 waves.
// ---------------------------------------------------------------------------
__global__ __launch_bounds__(128) void out_proj_kernel(
    const __bf16* __restrict__ ctx, const __bf16* __restrict__ wvb,
    const float* __restrict__ xn, float* __restrict__ out) {
  const int lane = threadIdx.x & 31;
  const int w    = threadIdx.x >> 5;
  const int m0   = blockIdx.x * 32;
  const int o0   = blockIdx.y * 128 + w * 32;

  const __bf16* a0base = ctx + (size_t)m0 * HD;
  const __bf16* a1base = a0base + (size_t)16 * HD;
  const __bf16* b0base = wvb + (size_t)o0 * HD;  // Wv (256,2048): B col n = row o0+n
  const __bf16* b1base = b0base + (size_t)16 * HD;

  v8f acc00 = {}, acc01 = {}, acc10 = {}, acc11 = {};
  for (int k = 0; k < HD / 32; ++k) {
    const int kb_ = k * 32;
    v16bf a0 = load_a16x32(a0base, HD, kb_, lane);
    v16bf a1 = load_a16x32(a1base, HD, kb_, lane);
    v16bf b0 = load_b32x16(b0base, HD, kb_, lane);
    v16bf b1 = load_b32x16(b1base, HD, kb_, lane);
    acc00 = wmma_bf16(a0, b0, acc00);
    acc01 = wmma_bf16(a0, b1, acc01);
    acc10 = wmma_bf16(a1, b0, acc10);
    acc11 = wmma_bf16(a1, b1, acc11);
  }

  const int n = lane & 15, hh = lane >> 4;
#pragma unroll
  for (int r = 0; r < 8; ++r) {
    const size_t r0 = (size_t)(m0 + r + hh * 8) * DIM;
    const size_t r1 = (size_t)(m0 + 16 + r + hh * 8) * DIM;
    out[r0 + o0 + n]      = acc00[r] + xn[r0 + o0 + n];
    out[r0 + o0 + 16 + n] = acc01[r] + xn[r0 + o0 + 16 + n];
    out[r1 + o0 + n]      = acc10[r] + xn[r1 + o0 + n];
    out[r1 + o0 + 16 + n] = acc11[r] + xn[r1 + o0 + 16 + n];
  }
}

// ---------------------------------------------------------------------------
// Host-side launcher.
// ---------------------------------------------------------------------------
extern "C" void kernel_launch(void* const* d_in, const int* in_sizes, int n_in,
                              void* d_out, int out_size, void* d_ws,
                              size_t ws_size, hipStream_t stream) {
  const float* x     = (const float*)d_in[0];
  const float* Wq    = (const float*)d_in[1];
  const float* Wk    = (const float*)d_in[2];
  const float* Wv    = (const float*)d_in[3];
  const float* gamma = (const float*)d_in[4];
  const float* beta  = (const float*)d_in[5];

  float* out   = (float*)d_out;
  float* probs = out + (size_t)BS * SEQ * DIM;   // tuple order: (out, probs)

  char* ws = (char*)d_ws;
  size_t off = 0;
  float*  xn_f32 = (float*)(ws + off);  off += (size_t)BS * SEQ * DIM * 4;
  __bf16* xn_bf  = (__bf16*)(ws + off); off += (size_t)BS * SEQ * DIM * 2;
  __bf16* xnT_bf = (__bf16*)(ws + off); off += (size_t)BS * DIM * SEQ * 2;
  __bf16* wq_bf  = (__bf16*)(ws + off); off += (size_t)HD * DIM * 2;
  __bf16* wk_bf  = (__bf16*)(ws + off); off += (size_t)HD * DIM * 2;
  __bf16* wv_bf  = (__bf16*)(ws + off); off += (size_t)DIM * HD * 2;
  __bf16* q_bf   = (__bf16*)(ws + off); off += (size_t)BS * SEQ * HD * 2;
  __bf16* k_bf   = (__bf16*)(ws + off); off += (size_t)BS * SEQ * HD * 2;
  __bf16* ctx_bf = (__bf16*)(ws + off); off += (size_t)BS * SEQ * HD * 2;
  // total: 227 MB of d_ws

  // K1: layernorm over 16384 token rows
  ln_kernel<<<dim3(BS * SEQ), dim3(DIM), 0, stream>>>(
      x, gamma, beta, xn_f32, xn_bf, xnT_bf);

  // K2: weight conversion
  wcvt_kernel<<<dim3((HD * DIM) / 256), dim3(256), 0, stream>>>(
      Wq, Wk, Wv, wq_bf, wk_bf, wv_bf);

  // K3: Q and K projections (32x32 per wave)
  proj_qk_kernel<<<dim3((BS * SEQ) / 32, HD / 128, 2), dim3(128), 0, stream>>>(
      xn_bf, wq_bf, wk_bf, q_bf, k_bf);

  // K4: attention (scores + softmax + probs + context)
  const size_t attn_lds = (size_t)16 * SEQ * 4 + 16 * SEQ * 2 + 128 * 4; // 98816 B
  attn_kernel<<<dim3(SEQ / 16, HEADS, BS), dim3(128), attn_lds, stream>>>(
      q_bf, k_bf, xnT_bf, ctx_bf, probs);

  // K5: output projection + residual (32x32 per wave)
  out_proj_kernel<<<dim3((BS * SEQ) / 32, DIM / 128), dim3(128), 0, stream>>>(
      ctx_bf, wv_bf, xn_f32, out);
}
